// VQVAE_2D_41128606826945
// MI455X (gfx1250) — matrix operand order repaired and checked by reading, not compiled, not run on hardware
//
#include <hip/hip_runtime.h>
#include <hip/hip_bf16.h>
#include <stdint.h>

typedef __attribute__((ext_vector_type(16))) _Float16 v16h;
typedef __attribute__((ext_vector_type(8)))  _Float16 v8h;
typedef __attribute__((ext_vector_type(8)))  float    v8f;
typedef unsigned int u32x4 __attribute__((ext_vector_type(4)));
typedef unsigned int u32x8 __attribute__((ext_vector_type(8)));

// ---------------------------------------------------------------------------
// Implicit-GEMM conv on NHWC f16 activations (channels padded to 32, zeros).
// GEMM K ordering: k = (kh*KW + kw)*Cinp + ci  ->  every 32-wide K chunk has a
// fixed window cell, so A fragments are two contiguous 16B runs (b128 loads).
//
// Weights are packed TRANSPOSED [Npad][Ktot] (n-major).  Each block owns a
// 32-column N slab; the whole 32 x Ktot f16 slab (<=128KB of the 320KB/WGP
// LDS) is pulled in by ONE Tensor-Data-Mover op issued by wave 0, so the K
// loop has no LDS staging and no barriers at all.
//
// Block = 128 threads = 4 wave32s; block tile 256(M) x 32(N); each wave runs
// 4x2 = 8 WMMAs per K chunk.
// ---------------------------------------------------------------------------
struct ConvDesc {
  int Cinp, H, W;                 // input NHWC, Cinp % 32 == 0 (zero padded)
  int Cout, Npad, OH, OW;         // Npad = roundup(Cout, 32)
  int KH, KW, stride, pad, ldil;  // ldil=2 => ConvTranspose (lhs dilation 2)
  int M, Ktot;                    // M = 16*OH*OW, Ktot = KH*KW*Cinp
  int relu, outF32;
};

__device__ __forceinline__ v8f wmma_f16(const v16h& a, const v16h& b, const v8f& c) {
  return __builtin_amdgcn_wmma_f32_16x16x32_f16(false, a, false, b, (short)0, c,
                                                false, false);
}

__global__ __launch_bounds__(128)
void conv_wmma_kernel(ConvDesc d,
                      const _Float16* __restrict__ X,
                      const _Float16* __restrict__ WtT,   // [Npad][Ktot] n-major
                      const float*    __restrict__ bias,
                      const _Float16* __restrict__ res,
                      _Float16* __restrict__ Yh,
                      float*    __restrict__ Yf) {
  extern __shared__ _Float16 BsT[];   // [32][Ktot] n-major weight slab

  const int tid  = threadIdx.x;
  const int lane = tid & 31;
  const int half = lane >> 4;
  const int l16  = lane & 15;
  const int wave = __builtin_amdgcn_readfirstlane(tid >> 5);   // uniform scalar

  const int n0 = blockIdx.y * 32;
  const int m0 = blockIdx.x * 256 + wave * 64;

  // ---- TDM: one tensor_load_to_lds pulls the 32 x Ktot weight slab ----
  if (wave == 0) {
    const unsigned Ktot = (unsigned)d.Ktot;
    uint64_t gaddr = (uint64_t)(uintptr_t)(WtT + (size_t)n0 * d.Ktot);
    // D# group0: count=1 | lds_addr=0 | global_addr | type=2
    u32x4 g0 = { 1u,                       // count = 1 (valid descriptor)
                 0u,                       // lds_addr (bytes) = 0
                 (unsigned)(gaddr & 0xffffffffu),
                 (unsigned)((gaddr >> 32) & 0x01ffffffu) | (2u << 30) };
    // D# group1: data_size=2B, tensor 32 x Ktot, tile 32 x Ktot, stride Ktot
    u32x8 g1 = { 1u << 16,                              // data_size = 1 (2B)
                 (Ktot & 0xffffu) << 16,                // tensor_dim0[15:0]
                 (Ktot >> 16) | (32u << 16),            // dim0 hi | tensor_dim1
                 (Ktot << 16),                          // tile_dim0 (bits 127:112)
                 32u,                                   // tile_dim1
                 Ktot,                                  // tensor_dim0_stride lo
                 0u, 0u };
    u32x4 gz = { 0u, 0u, 0u, 0u };
    asm volatile("tensor_load_to_lds %0, %1, %2, %3"
                 :
                 : "s"(g0), "s"(g1), "s"(gz), "s"(gz)
                 : "memory");
    __builtin_amdgcn_s_wait_tensorcnt(0);
  }
  __syncthreads();

  const int OW = d.OW;
  const int HW = d.OH * OW;

  // the four A rows (M-fragments) owned by this lane
  int bA[4], ohA[4], owA[4];
#pragma unroll
  for (int mf = 0; mf < 4; ++mf) {
    int row = m0 + mf * 16 + l16;
    int bb  = row / HW;
    int rr  = row - bb * HW;
    int oh  = rr / OW;
    bA[mf] = bb; ohA[mf] = oh; owA[mf] = rr - oh * OW;
  }

  v8f acc[8] = {};                    // [mf*2 + nf]
  const int kbA = half * 8;           // A K-base per WMMA f16 layout
  const int kbB = half * 8;           // B K-base

  const int nCells = d.KH * d.KW;
  int kc = 0;
  for (int cell = 0; cell < nCells; ++cell) {
    const int kh = cell / d.KW;
    const int kw = cell - kh * d.KW;

    const _Float16* baseA[4];
    bool validA[4];
#pragma unroll
    for (int mf = 0; mf < 4; ++mf) {
      int ih = ohA[mf] * d.stride - d.pad + kh;
      int iw = owA[mf] * d.stride - d.pad + kw;
      bool ok = true;
      if (d.ldil == 2) {              // transposed conv: input dilated by 2
        ok = (ih >= 0) && (iw >= 0) && !(ih & 1) && !(iw & 1);
        ih >>= 1; iw >>= 1;
      }
      ok = ok && ih >= 0 && ih < d.H && iw >= 0 && iw < d.W;
      if (!ok) { ih = 0; iw = 0; }
      validA[mf] = ok;
      baseA[mf]  = X + ((size_t)(bA[mf] * d.H + ih) * d.W + iw) * d.Cinp;
    }

    for (int cib = 0; cib < d.Cinp; cib += 32, kc += 32) {
      // ---- B fragments straight from the TDM-resident slab (2x b128) ----
      v16h bfr[2];
#pragma unroll
      for (int nf = 0; nf < 2; ++nf) {
        const _Float16* p = &BsT[(size_t)(nf * 16 + l16) * d.Ktot + kc + kbB];
        v8h lo = *(const v8h*)p;
        v8h hi = *(const v8h*)(p + 16);
#pragma unroll
        for (int e = 0; e < 8; ++e) { bfr[nf][e] = lo[e]; bfr[nf][8 + e] = hi[e]; }
      }

      // ---- A fragments: two contiguous b128 runs per fragment ----
#pragma unroll
      for (int mf = 0; mf < 4; ++mf) {
        v16h afr;
        if (validA[mf]) {
          const _Float16* p = baseA[mf] + cib + kbA;
          v8h lo = *(const v8h*)p;
          v8h hi = *(const v8h*)(p + 16);
#pragma unroll
          for (int e = 0; e < 8; ++e) { afr[e] = lo[e]; afr[8 + e] = hi[e]; }
        } else {
#pragma unroll
          for (int e = 0; e < 16; ++e) afr[e] = (_Float16)0.f;
        }
        acc[mf * 2 + 0] = wmma_f16(afr, bfr[0], acc[mf * 2 + 0]);
        acc[mf * 2 + 1] = wmma_f16(afr, bfr[1], acc[mf * 2 + 1]);
      }
    }
  }

  // ---- fused epilogue: bias -> ReLU -> residual add -> store ----
#pragma unroll
  for (int nf = 0; nf < 2; ++nf) {
    const int col   = n0 + nf * 16 + l16;
    const bool live = col < d.Cout;
    const float bv  = (bias && live) ? bias[col] : 0.f;
#pragma unroll
    for (int mf = 0; mf < 4; ++mf) {
      v8f a = acc[mf * 2 + nf];
#pragma unroll
      for (int r = 0; r < 8; ++r) {
        int m = m0 + mf * 16 + r + 8 * half;
        float v = a[r] + bv;
        if (d.relu) v = fmaxf(v, 0.f);
        if (d.outF32) {                              // final layer -> NCHW f32
          if (live) {
            int bb = m / HW; int rr = m - bb * HW;
            int oh = rr / OW; int ow = rr - oh * OW;
            Yf[((size_t)(bb * d.Cout + col) * d.OH + oh) * OW + ow] = v;
          }
        } else {                                     // NHWC f16 (padded cols = 0)
          size_t off = (size_t)m * d.Npad + col;
          if (!live) v = 0.f;
          else if (res) v += (float)res[off];
          Yh[off] = (_Float16)v;
        }
      }
    }
  }
}

// ---------------------------------------------------------------------------
// Weight pack: f32 OIHW (or torch ConvT (Cin,Cout,kh,kw) with spatial flip)
// -> f16 TRANSPOSED [n][k], k = (kh*KW+kw)*Cinp + ci, zero padded.
// ---------------------------------------------------------------------------
__global__ void pack_w_kernel(const float* __restrict__ src, _Float16* __restrict__ dst,
                              int Cout, int Cin, int Cinp, int KH, int KW,
                              int Npad, int Ktot, int transposed) {
  int idx = blockIdx.x * blockDim.x + threadIdx.x;
  int total = Npad * Ktot;
  if (idx >= total) return;
  int n = idx / Ktot, k = idx - (idx / Ktot) * Ktot;
  int cell = k / Cinp, ci = k - cell * Cinp;
  int kh = cell / KW, kw = cell - kh * KW;
  float v = 0.f;
  if (n < Cout && ci < Cin) {
    if (!transposed)
      v = src[((size_t)(n * Cin + ci) * KH + kh) * KW + kw];
    else
      v = src[((size_t)(ci * Cout + n) * KH + (KH - 1 - kh)) * KW + (KW - 1 - kw)];
  }
  dst[idx] = (_Float16)v;
}

// input f32 NCHW (16,3,256,256) -> f16 NHWC padded to 32 channels (zeros)
__global__ void x_to_nhwc_kernel(const float* __restrict__ src, _Float16* __restrict__ dst) {
  int i = blockIdx.x * blockDim.x + threadIdx.x;
  if (i >= 16 * 256 * 256 * 32) return;
  int c = i & 31, p = i >> 5;              // p = (b*256 + h)*256 + w
  float v = 0.f;
  if (c < 3) {
    int w = p & 255, rest = p >> 8;
    int h = rest & 255, b = rest >> 8;
    v = src[(((size_t)b * 3 + c) * 256 + h) * 256 + w];
  }
  dst[i] = (_Float16)v;
}

__global__ void zero_loss_kernel(float* loss) { *loss = 0.f; }

// ---------------------------------------------------------------------------
// VQ: codebook 512x64 resident in LDS (64 KB of the 320 KB/WGP pool).
// z/q are NHWC -> each latent vector is one contiguous 128B row (b128 loads).
// ---------------------------------------------------------------------------
__global__ __launch_bounds__(256)
void vq_kernel(const _Float16* __restrict__ z, const float* __restrict__ emb,
               _Float16* __restrict__ q, float* __restrict__ loss) {
  extern __shared__ _Float16 E[];                 // 512 * 64 f16
  for (int i = threadIdx.x; i < 512 * 64; i += 256) E[i] = (_Float16)emb[i];
  __syncthreads();

  const int n = blockIdx.x * 256 + threadIdx.x;   // 0 .. 65535
  const _Float16* zp = z + (size_t)n * 64;

  float x[64];
#pragma unroll
  for (int ch = 0; ch < 8; ++ch) {
    v8h v = *(const v8h*)(zp + ch * 8);
#pragma unroll
    for (int e = 0; e < 8; ++e) x[ch * 8 + e] = (float)v[e];
  }

  float best = 3.4e38f;
  int bi = 0;
  for (int c = 0; c < 512; ++c) {
    const _Float16* e = E + c * 64;
    float s = 0.f;
#pragma unroll
    for (int d = 0; d < 64; ++d) {
      float dv = x[d] - (float)e[d];
      s = fmaf(dv, dv, s);
    }
    if (s < best) { best = s; bi = c; }
  }

  _Float16* qp = q + (size_t)n * 64;
  const _Float16* e = E + bi * 64;
  float ls = 0.f;
#pragma unroll
  for (int ch = 0; ch < 8; ++ch) {
    v8h v;
#pragma unroll
    for (int eo = 0; eo < 8; ++eo) {
      float ev = (float)e[ch * 8 + eo];
      v[eo] = (_Float16)ev;
      float dv = ev - x[ch * 8 + eo];
      ls = fmaf(dv, dv, ls);
    }
    *(v8h*)(qp + ch * 8) = v;
  }
  atomicAdd(loss, ls * (1.25f / (65536.f * 64.f)));
}

// ---------------------------------------------------------------------------
// Host side
// ---------------------------------------------------------------------------
static ConvDesc mkDesc(int Cin, int H, int W, int Cout, int OH, int OW,
                       int KH, int KW, int stride, int pad, int ldil,
                       int relu, int outF32) {
  ConvDesc d;
  d.Cinp = (Cin + 31) & ~31;
  d.H = H; d.W = W;
  d.Cout = Cout; d.Npad = (Cout + 31) & ~31;
  d.OH = OH; d.OW = OW;
  d.KH = KH; d.KW = KW; d.stride = stride; d.pad = pad; d.ldil = ldil;
  d.M = 16 * OH * OW;
  d.Ktot = KH * KW * d.Cinp;
  d.relu = relu; d.outF32 = outF32;
  return d;
}

static void launch_conv(hipStream_t s, const ConvDesc& d,
                        const _Float16* X, const _Float16* Wt,
                        const float* bias, const _Float16* res,
                        _Float16* Yh, float* Yf) {
  dim3 grid(d.M / 256, d.Npad / 32);
  size_t shmem = (size_t)32 * d.Ktot * sizeof(_Float16);   // <= 128 KB
  conv_wmma_kernel<<<grid, dim3(128), shmem, s>>>(d, X, Wt, bias, res, Yh, Yf);
}

static void launch_pack(hipStream_t s, const float* src, _Float16* dst,
                        int Cout, int Cin, int KH, int KW, int transposed) {
  int Cinp = (Cin + 31) & ~31;
  int Npad = (Cout + 31) & ~31;
  int Ktot = KH * KW * Cinp;
  int total = Npad * Ktot;
  pack_w_kernel<<<(total + 255) / 256, 256, 0, s>>>(src, dst, Cout, Cin, Cinp,
                                                    KH, KW, Npad, Ktot, transposed);
}

extern "C" void kernel_launch(void* const* d_in, const int* in_sizes, int n_in,
                              void* d_out, int out_size, void* d_ws, size_t ws_size,
                              hipStream_t stream) {
  (void)in_sizes; (void)n_in; (void)out_size; (void)ws_size;

  const float* x    = (const float*)d_in[0];
  const float* emb  = (const float*)d_in[1];
  const float* ew1  = (const float*)d_in[2];
  const float* eb1  = (const float*)d_in[3];
  const float* ew2  = (const float*)d_in[4];
  const float* eb2  = (const float*)d_in[5];
  const float* er1a = (const float*)d_in[6];
  const float* er1b = (const float*)d_in[7];
  const float* er2a = (const float*)d_in[8];
  const float* er2b = (const float*)d_in[9];
  const float* ew3  = (const float*)d_in[10];
  const float* eb3  = (const float*)d_in[11];
  const float* dw1  = (const float*)d_in[12];
  const float* db1  = (const float*)d_in[13];
  const float* dr1a = (const float*)d_in[14];
  const float* dr1b = (const float*)d_in[15];
  const float* dr2a = (const float*)d_in[16];
  const float* dr2b = (const float*)d_in[17];
  const float* dtw1 = (const float*)d_in[18];
  const float* dtb1 = (const float*)d_in[19];
  const float* dtw2 = (const float*)d_in[20];
  const float* dtb2 = (const float*)d_in[21];

  uintptr_t cur = (uintptr_t)d_ws;
  auto alloc = [&](size_t elems) -> _Float16* {
    cur = (cur + 255) & ~(uintptr_t)255;
    _Float16* p = (_Float16*)cur;
    cur += elems * sizeof(_Float16);
    return p;
  };

  // NHWC activations, channels padded to 32
  _Float16* xh  = alloc((size_t)16 * 256 * 256 * 32);
  _Float16* h1  = alloc((size_t)16 * 128 * 128 * 64);   // reused for convT1 out
  _Float16* h2a = alloc((size_t)16 * 64 * 64 * 128);
  _Float16* h2b = alloc((size_t)16 * 64 * 64 * 128);
  _Float16* t   = alloc((size_t)16 * 64 * 64 * 32);
  _Float16* z   = alloc((size_t)16 * 64 * 64 * 64);
  _Float16* q   = alloc((size_t)16 * 64 * 64 * 64);

  _Float16* wc1   = alloc(16 * 32 * 64);
  _Float16* wc2   = alloc(16 * 64 * 128);
  _Float16* wr1a  = alloc(9 * 128 * 32);
  _Float16* wr1b  = alloc(1 * 32 * 128);
  _Float16* wr2a  = alloc(9 * 128 * 32);
  _Float16* wr2b  = alloc(1 * 32 * 128);
  _Float16* wc3   = alloc(9 * 128 * 64);
  _Float16* wd1   = alloc(9 * 64 * 128);
  _Float16* wdr1a = alloc(9 * 128 * 32);
  _Float16* wdr1b = alloc(1 * 32 * 128);
  _Float16* wdr2a = alloc(9 * 128 * 32);
  _Float16* wdr2b = alloc(1 * 32 * 128);
  _Float16* wt1   = alloc(16 * 128 * 64);
  _Float16* wt2   = alloc(16 * 64 * 32);

  float* out_img = (float*)d_out;
  float* loss    = (float*)d_out + (size_t)16 * 3 * 256 * 256;

  // ---- weight packing (transposed n-major for the TDM slab loads) ----
  launch_pack(stream, ew1,  wc1,  64, 3, 4, 4, 0);
  launch_pack(stream, ew2,  wc2,  128, 64, 4, 4, 0);
  launch_pack(stream, er1a, wr1a, 32, 128, 3, 3, 0);
  launch_pack(stream, er1b, wr1b, 128, 32, 1, 1, 0);
  launch_pack(stream, er2a, wr2a, 32, 128, 3, 3, 0);
  launch_pack(stream, er2b, wr2b, 128, 32, 1, 1, 0);
  launch_pack(stream, ew3,  wc3,  64, 128, 3, 3, 0);
  launch_pack(stream, dw1,  wd1,  128, 64, 3, 3, 0);
  launch_pack(stream, dr1a, wdr1a, 32, 128, 3, 3, 0);
  launch_pack(stream, dr1b, wdr1b, 128, 32, 1, 1, 0);
  launch_pack(stream, dr2a, wdr2a, 32, 128, 3, 3, 0);
  launch_pack(stream, dr2b, wdr2b, 128, 32, 1, 1, 0);
  launch_pack(stream, dtw1, wt1,  64, 128, 4, 4, 1);    // ConvT: flip + transpose
  launch_pack(stream, dtw2, wt2,  3, 64, 4, 4, 1);

  {
    int n = 16 * 256 * 256 * 32;
    x_to_nhwc_kernel<<<(n + 255) / 256, 256, 0, stream>>>(x, xh);
  }
  zero_loss_kernel<<<1, 1, 0, stream>>>(loss);

  // ---- encoder ----
  ConvDesc c1 = mkDesc(3, 256, 256, 64, 128, 128, 4, 4, 2, 1, 1, 1, 0);
  launch_conv(stream, c1, xh, wc1, eb1, nullptr, h1, nullptr);

  ConvDesc c2 = mkDesc(64, 128, 128, 128, 64, 64, 4, 4, 2, 1, 1, 1, 0);
  launch_conv(stream, c2, h1, wc2, eb2, nullptr, h2a, nullptr);

  ConvDesc r3 = mkDesc(128, 64, 64, 32, 64, 64, 3, 3, 1, 1, 1, 1, 0);   // 3x3 relu
  ConvDesc r1 = mkDesc(32, 64, 64, 128, 64, 64, 1, 1, 1, 0, 1, 1, 0);   // 1x1 relu+res

  launch_conv(stream, r3, h2a, wr1a, nullptr, nullptr, t, nullptr);
  launch_conv(stream, r1, t, wr1b, nullptr, h2a, h2b, nullptr);
  launch_conv(stream, r3, h2b, wr2a, nullptr, nullptr, t, nullptr);
  launch_conv(stream, r1, t, wr2b, nullptr, h2b, h2a, nullptr);

  ConvDesc c3 = mkDesc(128, 64, 64, 64, 64, 64, 3, 3, 1, 1, 1, 0, 0);
  launch_conv(stream, c3, h2a, wc3, eb3, nullptr, z, nullptr);

  // ---- vector quantization ----
  vq_kernel<<<256, 256, 512 * 64 * sizeof(_Float16), stream>>>(z, emb, q, loss);

  // ---- decoder ----
  ConvDesc dc = mkDesc(64, 64, 64, 128, 64, 64, 3, 3, 1, 1, 1, 0, 0);
  launch_conv(stream, dc, q, wd1, db1, nullptr, h2b, nullptr);

  launch_conv(stream, r3, h2b, wdr1a, nullptr, nullptr, t, nullptr);
  launch_conv(stream, r1, t, wdr1b, nullptr, h2b, h2a, nullptr);
  launch_conv(stream, r3, h2a, wdr2a, nullptr, nullptr, t, nullptr);
  launch_conv(stream, r1, t, wdr2b, nullptr, h2a, h2b, nullptr);

  ConvDesc ct1 = mkDesc(128, 64, 64, 64, 128, 128, 4, 4, 1, 2, 2, 1, 0);
  launch_conv(stream, ct1, h2b, wt1, dtb1, nullptr, h1, nullptr);

  ConvDesc ct2 = mkDesc(64, 128, 128, 3, 256, 256, 4, 4, 1, 2, 2, 1, 1);
  launch_conv(stream, ct2, h1, wt2, dtb2, nullptr, nullptr, out_img);
}
